// MyGaussianRasterizer_39135742001608
// MI455X (gfx1250) — matrix operand order repaired and checked by reading, not compiled, not run on hardware
//
#include <hip/hip_runtime.h>
#include <math.h>

typedef __attribute__((ext_vector_type(16))) _Float16 v16h;
typedef __attribute__((ext_vector_type(8)))  float    v8f;

#define IMGS    128
#define NGAUSS  2048
#define NFRAMES 2
#define BN      (NFRAMES * NGAUSS)
#define INFFAR  1000.0f

// workspace layout: float arrays of length BN at these slot indices
#define WS_UMX   0
#define WS_UMY   1
#define WS_UCA   2
#define WS_UCB   3
#define WS_UCC   4
#define WS_UDEP  5
#define WS_UOP   6
#define WS_UR    7
#define WS_UG    8
#define WS_UB    9
#define WS_UKEY  10
#define WS_SMX   11
#define WS_SMY   12
#define WS_SCA   13
#define WS_SCB   14
#define WS_SCC   15
#define WS_SDEP  16
#define WS_SOP   17
#define WS_HALF  18   // 4 half arrays (r,g,b,depth) of BN start here

// ---------------------------------------------------------------------------
// Kernel 1: per-gaussian projection, 2D covariance, conic, radii, sort keys
// ---------------------------------------------------------------------------
__global__ __launch_bounds__(256) void gs_preprocess(
    const float* __restrict__ xyz,    const float* __restrict__ colors,
    const float* __restrict__ opac,   const float* __restrict__ scales,
    const float* __restrict__ rots,   const float* __restrict__ camT,
    const float* __restrict__ camP,   float* __restrict__ ws,
    int* __restrict__ radii_out)
{
  int gid = blockIdx.x * blockDim.x + threadIdx.x;
  if (gid >= BN) return;
  int f = gid / NGAUSS;
  const float* T = camT + f * 16;   // cam_T row-major
  const float* P = camP + f * 16;   // cam_P row-major

  float x = xyz[gid * 3 + 0], y = xyz[gid * 3 + 1], z = xyz[gid * 3 + 2];

  // p_view[j] = dot(cam_T row j, [x,y,z,1])  (ph @ cam_T.T)
  float pv[4];
#pragma unroll
  for (int j = 0; j < 4; ++j)
    pv[j] = T[j * 4 + 0] * x + T[j * 4 + 1] * y + T[j * 4 + 2] * z + T[j * 4 + 3];
  // p_proj[j] = Σ_m cam_P[j][m] * pv[m]
  float pp[4];
#pragma unroll
  for (int j = 0; j < 4; ++j)
    pp[j] = P[j * 4 + 0] * pv[0] + P[j * 4 + 1] * pv[1] +
            P[j * 4 + 2] * pv[2] + P[j * 4 + 3] * pv[3];

  float depth = pv[2];
  float invw  = 1.0f / (pp[3] + 1e-7f);
  float mx = ((pp[0] * invw + 1.0f) * (float)IMGS - 1.0f) * 0.5f;
  float my = ((pp[1] * invw + 1.0f) * (float)IMGS - 1.0f) * 0.5f;

  float tanx = 1.0f / P[0];     // proj[0,0] = cam_P[0][0]
  float tany = 1.0f / P[5];
  float fx = (float)IMGS / (2.0f * tanx);
  float fy = (float)IMGS / (2.0f * tany);

  // quaternion (w,x,y,z) -> rotation
  float qw = rots[gid * 4 + 0], qx = rots[gid * 4 + 1];
  float qy = rots[gid * 4 + 2], qz = rots[gid * 4 + 3];
  float qn = rsqrtf(qw * qw + qx * qx + qy * qy + qz * qz);
  qw *= qn; qx *= qn; qy *= qn; qz *= qn;
  float R[3][3];
  R[0][0] = 1.f - 2.f * (qy * qy + qz * qz); R[0][1] = 2.f * (qx * qy - qw * qz); R[0][2] = 2.f * (qx * qz + qw * qy);
  R[1][0] = 2.f * (qx * qy + qw * qz); R[1][1] = 1.f - 2.f * (qx * qx + qz * qz); R[1][2] = 2.f * (qy * qz - qw * qx);
  R[2][0] = 2.f * (qx * qz - qw * qy); R[2][1] = 2.f * (qy * qz + qw * qx); R[2][2] = 1.f - 2.f * (qx * qx + qy * qy);

  float s3[3] = { scales[gid * 3 + 0], scales[gid * 3 + 1], scales[gid * 3 + 2] };
  float M3[3][3], Sig[3][3];
#pragma unroll
  for (int i = 0; i < 3; ++i)
#pragma unroll
    for (int j = 0; j < 3; ++j) M3[i][j] = R[i][j] * s3[j];
#pragma unroll
  for (int i = 0; i < 3; ++i)
#pragma unroll
    for (int k = 0; k < 3; ++k)
      Sig[i][k] = M3[i][0] * M3[k][0] + M3[i][1] * M3[k][1] + M3[i][2] * M3[k][2];

  float tz  = (depth > 0.2f) ? depth : 1.0f;
  float itz = 1.0f / tz;
  float lx = 1.3f * tanx, ly = 1.3f * tany;
  float txc = fminf(fmaxf(pv[0] * itz, -lx), lx) * tz;
  float tyc = fminf(fmaxf(pv[1] * itz, -ly), ly) * tz;
  float J00 = fx * itz, J02 = -fx * txc * itz * itz;
  float J11 = fy * itz, J12 = -fy * tyc * itz * itz;

  // Wr = cam_T[:3,:3];  T2 = J @ Wr
  float T20[3], T21[3];
#pragma unroll
  for (int k = 0; k < 3; ++k) {
    T20[k] = J00 * T[0 * 4 + k] + J02 * T[2 * 4 + k];
    T21[k] = J11 * T[1 * 4 + k] + J12 * T[2 * 4 + k];
  }
  float t0[3], t1[3];
#pragma unroll
  for (int k = 0; k < 3; ++k) {
    t0[k] = T20[0] * Sig[0][k] + T20[1] * Sig[1][k] + T20[2] * Sig[2][k];
    t1[k] = T21[0] * Sig[0][k] + T21[1] * Sig[1][k] + T21[2] * Sig[2][k];
  }
  float cov00 = t0[0] * T20[0] + t0[1] * T20[1] + t0[2] * T20[2];
  float cov01 = t0[0] * T21[0] + t0[1] * T21[1] + t0[2] * T21[2];
  float cov11 = t1[0] * T21[0] + t1[1] * T21[1] + t1[2] * T21[2];

  float a = cov00 + 0.3f, b = cov01, c = cov11 + 0.3f;
  float det = a * c - b * b;
  bool valid = (depth > 0.2f) && (det > 0.0f);
  float det_s = (det > 0.0f) ? det : 1.0f;
  float conA =  c / det_s, conB = -b / det_s, conC = a / det_s;
  float mid  = 0.5f * (a + c);
  float lam1 = mid + sqrtf(fmaxf(mid * mid - det, 0.1f));
  radii_out[gid] = valid ? (int)ceilf(3.0f * sqrtf(lam1)) : 0;

  ws[WS_UMX  * BN + gid] = mx;
  ws[WS_UMY  * BN + gid] = my;
  ws[WS_UCA  * BN + gid] = conA;
  ws[WS_UCB  * BN + gid] = conB;
  ws[WS_UCC  * BN + gid] = conC;
  ws[WS_UDEP * BN + gid] = depth;
  ws[WS_UOP  * BN + gid] = valid ? opac[gid] : 0.0f;
  ws[WS_UR   * BN + gid] = colors[gid * 3 + 0];
  ws[WS_UG   * BN + gid] = colors[gid * 3 + 1];
  ws[WS_UB   * BN + gid] = colors[gid * 3 + 2];
  ws[WS_UKEY * BN + gid] = valid ? depth : __builtin_huge_valf();
}

// ---------------------------------------------------------------------------
// Kernel 2: per-frame bitonic argsort of 2048 depth keys + gather sorted data
// ---------------------------------------------------------------------------
__global__ __launch_bounds__(1024) void gs_sort_gather(float* __restrict__ ws)
{
  __shared__ float skey[NGAUSS];
  __shared__ int   sidx[NGAUSS];
  int f = blockIdx.x, t = threadIdx.x;
  int fb = f * NGAUSS;
  const float* ukey = ws + WS_UKEY * BN + fb;

  skey[t] = ukey[t];                 sidx[t] = t;
  skey[t + 1024] = ukey[t + 1024];   sidx[t + 1024] = t + 1024;
  __syncthreads();

  for (int k = 2; k <= NGAUSS; k <<= 1) {
    for (int j = k >> 1; j > 0; j >>= 1) {
      int i1 = (t / j) * (2 * j) + (t % j);
      int i2 = i1 + j;
      bool up = ((i1 & k) == 0);
      float a = skey[i1], b = skey[i2];
      if (up ? (a > b) : (a < b)) {
        skey[i1] = b; skey[i2] = a;
        int ia = sidx[i1]; sidx[i1] = sidx[i2]; sidx[i2] = ia;
      }
      __syncthreads();
    }
  }

  _Float16* hb = (_Float16*)(ws + WS_HALF * BN);
  for (int s = t; s < NGAUSS; s += 1024) {
    int src = sidx[s];
    float dv = ws[WS_UDEP * BN + fb + src];
    ws[WS_SMX  * BN + fb + s] = ws[WS_UMX * BN + fb + src];
    ws[WS_SMY  * BN + fb + s] = ws[WS_UMY * BN + fb + src];
    ws[WS_SCA  * BN + fb + s] = ws[WS_UCA * BN + fb + src];
    ws[WS_SCB  * BN + fb + s] = ws[WS_UCB * BN + fb + src];
    ws[WS_SCC  * BN + fb + s] = ws[WS_UCC * BN + fb + src];
    ws[WS_SDEP * BN + fb + s] = dv;
    ws[WS_SOP  * BN + fb + s] = ws[WS_UOP * BN + fb + src];
    hb[0 * BN + fb + s] = (_Float16)ws[WS_UR * BN + fb + src];
    hb[1 * BN + fb + s] = (_Float16)ws[WS_UG * BN + fb + src];
    hb[2 * BN + fb + s] = (_Float16)ws[WS_UB * BN + fb + src];
    hb[3 * BN + fb + s] = (_Float16)dv;
  }
}

// ---------------------------------------------------------------------------
// Kernel 3: rasterize. Wave = 16 pixels; 32-gaussian chunks; WMMA accumulate
// ---------------------------------------------------------------------------
__device__ inline void ld8f(float* d, const float* s) {
  float4 a = *(const float4*)(s);
  float4 b = *(const float4*)(s + 4);
  d[0] = a.x; d[1] = a.y; d[2] = a.z; d[3] = a.w;
  d[4] = b.x; d[5] = b.y; d[6] = b.z; d[7] = b.w;
}

__global__ __launch_bounds__(256) void gs_raster(
    const float* __restrict__ ws, const float* __restrict__ bg,
    const float* __restrict__ mesh_depth, const float* __restrict__ mask_uv,
    float* __restrict__ out_img, float* __restrict__ out_dep,
    float* __restrict__ out_acc)
{
  __shared__ float tile[8][16][17];

  int f    = blockIdx.x >> 7;
  int tid  = threadIdx.x;
  int wave = tid >> 5, lane = tid & 31;
  int p    = lane & 15, hi = lane >> 4;
  int pix  = (blockIdx.x & 127) * 128 + wave * 16 + p;
  float px = (float)(pix & (IMGS - 1));
  float py = (float)(pix >> 7);
  int fb = f * NGAUSS;

  const float* s_mx  = ws + WS_SMX  * BN + fb;
  const float* s_my  = ws + WS_SMY  * BN + fb;
  const float* s_ca  = ws + WS_SCA  * BN + fb;
  const float* s_cb  = ws + WS_SCB  * BN + fb;
  const float* s_cc  = ws + WS_SCC  * BN + fb;
  const float* s_dp  = ws + WS_SDEP * BN + fb;
  const float* s_op  = ws + WS_SOP  * BN + fb;
  const _Float16* hbase = (const _Float16*)(ws + WS_HALF * BN);
  const _Float16* hR = hbase + 0 * BN + fb;
  const _Float16* hG = hbase + 1 * BN + fb;
  const _Float16* hB = hbase + 2 * BN + fb;
  const _Float16* hD = hbase + 3 * BN + fb;

  float dcond = (1.0f - mask_uv[f * IMGS * IMGS + pix]) * INFFAR +
                mesh_depth[f * IMGS * IMGS + pix];

  float Tt = 1.0f;                     // running transmittance (both lane halves)
  v8f cacc = {0.f, 0.f, 0.f, 0.f, 0.f, 0.f, 0.f, 0.f};

  for (int cblk = 0; cblk < NGAUSS / 32; ++cblk) {
    int base = cblk * 32;
    if (cblk + 1 < NGAUSS / 32) __builtin_prefetch(s_mx + base + 32, 0, 0);

    // --- alpha for this lane's 16 gaussians (A-layout K assignment) ---
    float al[16];
    float P0 = 1.0f, P1 = 1.0f;
#pragma unroll
    for (int g = 0; g < 2; ++g) {
      int gs = base + g * 16 + hi * 8;
      float vmx[8], vmy[8], vca[8], vcb[8], vcc[8], vop[8], vdp[8];
      ld8f(vmx, s_mx + gs); ld8f(vmy, s_my + gs);
      ld8f(vca, s_ca + gs); ld8f(vcb, s_cb + gs); ld8f(vcc, s_cc + gs);
      ld8f(vop, s_op + gs); ld8f(vdp, s_dp + gs);
      float Pg = 1.0f;
#pragma unroll
      for (int i = 0; i < 8; ++i) {
        float dx = vmx[i] - px, dy = vmy[i] - py;
        float pw = -0.5f * (vca[i] * dx * dx + vcc[i] * dy * dy) - vcb[i] * dx * dy;
        float a0 = vop[i] * __expf(fminf(pw, 0.0f));
        bool keep = (pw <= 0.0f) && (a0 >= (1.0f / 255.0f)) && (vdp[i] <= dcond);
        float a1 = keep ? fminf(a0, 0.99f) : 0.0f;
        al[g * 8 + i] = a1;
        Pg *= (1.0f - a1);
      }
      if (g == 0) P0 = Pg; else P1 = Pg;
    }

    // --- ordered exclusive transmittance across the lane pair ---
    float P0o = __shfl_xor(P0, 16, 32);
    float P1o = __shfl_xor(P1, 16, 32);
    float pre0 = hi ? (Tt * P0o) : Tt;                       // prefix before group0
    float pre1 = Tt * P0 * P0o * (hi ? P1o : 1.0f);          // prefix before group1
    Tt = Tt * P0 * P0o * P1 * P1o;

    // --- pack A fragment: w = alpha * exclusive_prefix, f16 ---
    v16h afrag;
    float run = pre0;
#pragma unroll
    for (int i = 0; i < 8; ++i) { afrag[i] = (_Float16)(al[i] * run); run *= (1.0f - al[i]); }
    run = pre1;
#pragma unroll
    for (int i = 0; i < 8; ++i) { afrag[8 + i] = (_Float16)(al[8 + i] * run); run *= (1.0f - al[8 + i]); }

    // --- pack B fragment: column n = p, K at base + 8*hi and base + 16 + 8*hi ---
    v16h bfrag;
    int kb = base + hi * 8;
    if (p < 4) {
      const _Float16* srcp = (p == 0) ? hR : (p == 1) ? hG : (p == 2) ? hB : hD;
      union { int4 v; _Float16 h[8]; } u0, u1;
      u0.v = *(const int4*)(srcp + kb);
      u1.v = *(const int4*)(srcp + kb + 16);
#pragma unroll
      for (int i = 0; i < 8; ++i) { bfrag[i] = u0.h[i]; bfrag[8 + i] = u1.h[i]; }
    } else {
      _Float16 cval = (p == 4) ? (_Float16)1.0f : (_Float16)0.0f;
#pragma unroll
      for (int i = 0; i < 16; ++i) bfrag[i] = cval;
    }

    // D[pixel, channel] += W[16x32] x Attr[32x16]
    cacc = __builtin_amdgcn_wmma_f32_16x16x32_f16(
        false, afrag, false, bfrag, (short)0, cacc, false, false);
  }

  // spill C/D fragment to LDS: lane (p,hi) holds col p, rows hi*8 + j
#pragma unroll
  for (int j = 0; j < 8; ++j) tile[wave][j + 8 * hi][p] = cacc[j];
  __syncthreads();

  if (hi == 0) {
    float r = tile[wave][p][0], g = tile[wave][p][1], b = tile[wave][p][2];
    float d = tile[wave][p][3], a = tile[wave][p][4];
    float b0 = bg[f * 3 + 0], b1 = bg[f * 3 + 1], b2 = bg[f * 3 + 2];
    int o = f * 3 * IMGS * IMGS;
    out_img[o + 0 * IMGS * IMGS + pix] = r + Tt * b0;
    out_img[o + 1 * IMGS * IMGS + pix] = g + Tt * b1;
    out_img[o + 2 * IMGS * IMGS + pix] = b + Tt * b2;
    out_dep[f * IMGS * IMGS + pix] = d;
    out_acc[f * IMGS * IMGS + pix] = a;
  }
}

// ---------------------------------------------------------------------------
extern "C" void kernel_launch(void* const* d_in, const int* in_sizes, int n_in,
                              void* d_out, int out_size, void* d_ws, size_t ws_size,
                              hipStream_t stream) {
  const float* xyz        = (const float*)d_in[0];
  const float* colors     = (const float*)d_in[1];
  const float* opac       = (const float*)d_in[2];
  const float* scales     = (const float*)d_in[3];
  const float* rots       = (const float*)d_in[4];
  const float* background = (const float*)d_in[5];
  const float* camT       = (const float*)d_in[6];
  const float* camP       = (const float*)d_in[7];
  const float* mesh_depth = (const float*)d_in[8];
  const float* mask_uv    = (const float*)d_in[9];

  float* out      = (float*)d_out;
  float* out_img  = out;                         // [2,3,128,128] = 98304
  float* out_dep  = out + 98304;                 // [2,1,128,128] = 32768
  float* out_acc  = out + 131072;                // [2,1,128,128] = 32768
  int*   out_rad  = (int*)(out + 163840);        // [2,2048] int32
  float* ws = (float*)d_ws;

  gs_preprocess<<<BN / 256, 256, 0, stream>>>(xyz, colors, opac, scales, rots,
                                              camT, camP, ws, out_rad);
  gs_sort_gather<<<NFRAMES, 1024, 0, stream>>>(ws);
  gs_raster<<<NFRAMES * 128, 256, 0, stream>>>(ws, background, mesh_depth,
                                               mask_uv, out_img, out_dep, out_acc);
}
